// LSTMCell_90323162235593
// MI455X (gfx1250) — matrix operand order repaired and checked by reading, not compiled, not run on hardware
//
#include <hip/hip_runtime.h>
#include <hip/hip_bf16.h>

#define B_SZ 8192
#define D_SZ 1024
#define H_SZ 1024
#define K_TOT 2048              // concat [x | h] along K
#define BLOCK_M 128
#define BLOCK_N 64              // per gate
#define NGATE 4
#define BROWS (NGATE * BLOCK_N) // 256 W-rows per block
#define KSTEP 32
#define NSTAGES (K_TOT / KSTEP) // 64
#define THREADS 256

#define AS1 __attribute__((address_space(1)))
#define AS3 __attribute__((address_space(3)))

// Async global->LDS path (CDNA5 GLOBAL_LOAD_ASYNC_TO_LDS_B128, ASYNCcnt-tracked).
#if defined(__HIP_DEVICE_COMPILE__) && __has_builtin(__builtin_amdgcn_global_load_async_to_lds_b128)
#define USE_ASYNC 1
#else
#define USE_ASYNC 0
#endif

typedef __attribute__((ext_vector_type(16))) __bf16 v16bf;
typedef __attribute__((ext_vector_type(8)))  float  v8f;
typedef __attribute__((ext_vector_type(4)))  int    i4v;   // pointee type the async builtin expects

union V8F   { v8f   v; float f[8]; };
union AFrag { v16bf v; uint4 q[2]; };

__device__ __forceinline__ float sigmoid_(float x) { return 1.0f / (1.0f + __expf(-x)); }
__device__ __forceinline__ float tanh_(float x)    { return 2.0f / (1.0f + __expf(-2.0f * x)) - 1.0f; }

__device__ __forceinline__ void wait_async0() {
#if USE_ASYNC
#if __has_builtin(__builtin_amdgcn_s_wait_asynccnt)
    __builtin_amdgcn_s_wait_asynccnt(0);
#else
    asm volatile("s_wait_asynccnt 0" ::: "memory");
#endif
#endif
}

// ---------------- f32 -> bf16 pre-conversion (one pass, 8 elems/thread) ----------------
__global__ __launch_bounds__(THREADS) void convert_f32_bf16(
    const float* __restrict__ s, __bf16* __restrict__ d, int n)
{
    int i = (blockIdx.x * THREADS + threadIdx.x) * 8;
    if (i >= n) return;
    float4 a = *(const float4*)(s + i);
    float4 b = *(const float4*)(s + i + 4);
    union { __bf16 h[8]; uint4 q; } u;
    u.h[0] = (__bf16)a.x; u.h[1] = (__bf16)a.y; u.h[2] = (__bf16)a.z; u.h[3] = (__bf16)a.w;
    u.h[4] = (__bf16)b.x; u.h[5] = (__bf16)b.y; u.h[6] = (__bf16)b.z; u.h[7] = (__bf16)b.w;
    *(uint4*)(d + i) = u.q;
}

// ---------------- per-stage source selection: stages 0..31 = x*Wx, 32..63 = h*Wh ----------------
struct StageSrc { const __bf16* As; const __bf16* Bs; int kl; };
__device__ __forceinline__ StageSrc stage_src(int s,
    const __bf16* xb, const __bf16* hb, const __bf16* wxb, const __bf16* whb)
{
    StageSrc r;
    if (s < NSTAGES / 2) { r.As = xb; r.Bs = wxb; r.kl = s * KSTEP; }
    else                 { r.As = hb; r.Bs = whb; r.kl = (s - NSTAGES / 2) * KSTEP; }
    return r;
}

#if USE_ASYNC
// issue 6 async 16B copies per thread: A tile 128x32, B tile 256x32
__device__ __forceinline__ void issue_stage_async(int s, int tid, int m_base, int n_base,
    const __bf16* xb, const __bf16* hb, const __bf16* wxb, const __bf16* whb,
    __bf16* dA, __bf16* dB)
{
    StageSrc ss = stage_src(s, xb, hb, wxb, whb);
    #pragma unroll
    for (int i = 0; i < 2; ++i) {
        int c = tid + i * THREADS;
        int row = c >> 2, cir = c & 3;
        const __bf16* g = ss.As + (size_t)(m_base + row) * D_SZ + ss.kl + cir * 8;
        __bf16* l = dA + row * KSTEP + cir * 8;
        __builtin_amdgcn_global_load_async_to_lds_b128(
            (AS1 i4v*)g, (AS3 i4v*)l, 0, 0);
    }
    #pragma unroll
    for (int i = 0; i < 4; ++i) {
        int c = tid + i * THREADS;
        int row = c >> 2, cir = c & 3;
        int gq = row >> 6, j = row & 63;
        const __bf16* g = ss.Bs + (size_t)(gq * H_SZ + n_base + j) * D_SZ + ss.kl + cir * 8;
        __bf16* l = dB + row * KSTEP + cir * 8;
        __builtin_amdgcn_global_load_async_to_lds_b128(
            (AS1 i4v*)g, (AS3 i4v*)l, 0, 0);
    }
}
#endif

// ---------------- fallback: global -> register prefetch for one K-stage ----------------
__device__ __forceinline__ void load_stage(int s, int tid, int m_base, int n_base,
    const __bf16* __restrict__ xb, const __bf16* __restrict__ hb,
    const __bf16* __restrict__ wxb, const __bf16* __restrict__ whb,
    uint4* pA, uint4* pB)
{
    StageSrc ss = stage_src(s, xb, hb, wxb, whb);
    #pragma unroll
    for (int i = 0; i < 2; ++i) {
        int c = tid + i * THREADS;
        int row = c >> 2, cir = c & 3;
        pA[i] = *(const uint4*)(ss.As + (size_t)(m_base + row) * D_SZ + ss.kl + cir * 8);
    }
    #pragma unroll
    for (int i = 0; i < 4; ++i) {
        int c = tid + i * THREADS;
        int row = c >> 2, cir = c & 3;
        int g = row >> 6, j = row & 63;
        pB[i] = *(const uint4*)(ss.Bs + (size_t)(g * H_SZ + n_base + j) * D_SZ + ss.kl + cir * 8);
    }
}

__device__ __forceinline__ void store_stage(int tid, __bf16* dA, __bf16* dB,
                                            const uint4* pA, const uint4* pB)
{
    #pragma unroll
    for (int i = 0; i < 2; ++i) {
        int c = tid + i * THREADS;
        int row = c >> 2, cir = c & 3;
        *(uint4*)(dA + row * KSTEP + cir * 8) = pA[i];
    }
    #pragma unroll
    for (int i = 0; i < 4; ++i) {
        int c = tid + i * THREADS;
        int row = c >> 2, cir = c & 3;
        *(uint4*)(dB + row * KSTEP + cir * 8) = pB[i];
    }
}

// ---------------- fused LSTM-cell GEMM (bf16 WMMA, f32 acc) ----------------
// No waves_per_eu cap: with the stage loop pinned at unroll 1, natural pressure is
// ~215 VGPRs (4 waves/EU) and the allocator keeps multiple B fragments in flight.
__global__ __launch_bounds__(THREADS)
void lstm_cell_wmma(
    const __bf16* __restrict__ xb,  const __bf16* __restrict__ hb,
    const __bf16* __restrict__ wxb, const __bf16* __restrict__ whb,
    const float* __restrict__ bx,   const float* __restrict__ bh,
    const float* __restrict__ c1,
    float* __restrict__ outh, float* __restrict__ outc)
{
    __shared__ __bf16 lA[2][BLOCK_M * KSTEP]; // 16 KB
    __shared__ __bf16 lB[2][BROWS * KSTEP];   // 32 KB

    const int tid  = threadIdx.x;
    const int lane = tid & 31;
    const int wave = tid >> 5;
    const int half = lane >> 4;   // K/M half select per WMMA layout
    const int lm   = lane & 15;
    const int m_base = blockIdx.x * BLOCK_M;
    const int n_base = blockIdx.y * BLOCK_N;

    V8F acc[16]; // [gate][n-tile] : 4 x 4 tiles of 16x16 f32
    #pragma unroll
    for (int i = 0; i < 16; ++i)
        #pragma unroll
        for (int j = 0; j < 8; ++j) acc[i].f[j] = 0.0f;

    // prologue: stage 0 into buffer 0
#if USE_ASYNC
    issue_stage_async(0, tid, m_base, n_base, xb, hb, wxb, whb, lA[0], lB[0]);
    wait_async0();
#else
    {
        uint4 pA[2], pB[4];
        load_stage(0, tid, m_base, n_base, xb, hb, wxb, whb, pA, pB);
        store_stage(tid, lA[0], lB[0], pA, pB);
    }
#endif
    __syncthreads();

    // manually double-buffered; compiler unrolling only inflates register pressure
    #pragma unroll 1
    for (int s = 0; s < NSTAGES; ++s) {
        const int cur = s & 1;

#if USE_ASYNC
        if (s + 1 < NSTAGES)
            issue_stage_async(s + 1, tid, m_base, n_base, xb, hb, wxb, whb,
                              lA[cur ^ 1], lB[cur ^ 1]);
#else
        uint4 nA[2], nB[4];
        if (s + 1 < NSTAGES)
            load_stage(s + 1, tid, m_base, n_base, xb, hb, wxb, whb, nA, nB);
#endif

        // A fragment: lane lm = row, contiguous k = half*16 .. half*16+15 (matches B)
        AFrag a;
        const __bf16* arow = &lA[cur][(wave * 16 + lm) * KSTEP];
        a.q[0] = *(const uint4*)(arow + half * 16);
        a.q[1] = *(const uint4*)(arow + half * 16 + 8);

        // 16 tiles (4 gates x 4 n-tiles), two B-fragment buffers interleaved so loads
        // can overlap WMMA issue (in-order DScnt wait <= 2 instead of 0).
        const __bf16* lbase = &lB[cur][lm * KSTEP + half * 16];
        auto bfrag = [&](int i) -> v16bf {
            const int q = i >> 2, t = i & 3;
            return *(const v16bf*)(lbase + (q * BLOCK_N + t * 16) * KSTEP);
        };
        v16bf bA = bfrag(0);
        v16bf bB = bfrag(1);
        #pragma unroll
        for (int i = 0; i < 16; i += 2) {
            acc[i].v = __builtin_amdgcn_wmma_f32_16x16x32_bf16(
                false, a.v, false, bA, (short)0, acc[i].v, false, false);
            if (i + 2 < 16) bA = bfrag(i + 2);
            acc[i + 1].v = __builtin_amdgcn_wmma_f32_16x16x32_bf16(
                false, a.v, false, bB, (short)0, acc[i + 1].v, false, false);
            if (i + 3 < 16) bB = bfrag(i + 3);
        }

#if USE_ASYNC
        wait_async0();           // next-stage LDS writes landed (per-wave), then publish
#else
        if (s + 1 < NSTAGES)
            store_stage(tid, lA[cur ^ 1], lB[cur ^ 1], nA, nB);
#endif
        __syncthreads();
    }

    // fused gate epilogue: i/f/o/c for the same (row,col) live in registers
    // acc index i = q*4 + t
    #pragma unroll
    for (int t = 0; t < 4; ++t) {
        const int col = n_base + t * 16 + lm;
        float bsum[NGATE];
        #pragma unroll
        for (int q = 0; q < NGATE; ++q)
            bsum[q] = bx[q * H_SZ + col] + bh[q * H_SZ + col];
        #pragma unroll
        for (int v = 0; v < 8; ++v) {
            const int row = m_base + wave * 16 + half * 8 + v; // C/D layout: lanes16-31 -> M+8
            float gi = acc[0 * 4 + t].f[v] + bsum[0];
            float gf = acc[1 * 4 + t].f[v] + bsum[1];
            float go = acc[2 * 4 + t].f[v] + bsum[2];
            float gc = acc[3 * 4 + t].f[v] + bsum[3];
            float i_ = sigmoid_(gi);
            float f_ = sigmoid_(gf);
            float o_ = sigmoid_(go);
            float c_ = tanh_(gc);
            float cp = c1[(size_t)row * H_SZ + col];
            float cn = sigmoid_(f_ * cp + i_ * c_);   // reference quirk kept
            float hn = tanh_(cn) * o_;
            outh[(size_t)row * H_SZ + col] = hn;
            outc[(size_t)row * H_SZ + col] = cn;
        }
    }
}

extern "C" void kernel_launch(void* const* d_in, const int* in_sizes, int n_in,
                              void* d_out, int out_size, void* d_ws, size_t ws_size,
                              hipStream_t stream)
{
    const float* x  = (const float*)d_in[0];
    const float* h  = (const float*)d_in[1];
    const float* c1 = (const float*)d_in[2];
    const float* Wx = (const float*)d_in[3];
    const float* bx = (const float*)d_in[4];
    const float* Wh = (const float*)d_in[5];
    const float* bh = (const float*)d_in[6];

    // bf16 staging in workspace: x | h | Wx | Wh  (48 MB total)
    __bf16* xb  = (__bf16*)d_ws;
    __bf16* hb  = xb  + (size_t)B_SZ * D_SZ;
    __bf16* wxb = hb  + (size_t)B_SZ * D_SZ;
    __bf16* whb = wxb + (size_t)NGATE * H_SZ * D_SZ;

    const int nx = B_SZ * D_SZ;          // 8388608
    const int nw = NGATE * H_SZ * D_SZ;  // 4194304
    convert_f32_bf16<<<(nx / 8 + THREADS - 1) / THREADS, THREADS, 0, stream>>>(x,  xb,  nx);
    convert_f32_bf16<<<(nx / 8 + THREADS - 1) / THREADS, THREADS, 0, stream>>>(h,  hb,  nx);
    convert_f32_bf16<<<(nw / 8 + THREADS - 1) / THREADS, THREADS, 0, stream>>>(Wx, wxb, nw);
    convert_f32_bf16<<<(nw / 8 + THREADS - 1) / THREADS, THREADS, 0, stream>>>(Wh, whb, nw);

    float* outh = (float*)d_out;
    float* outc = outh + (size_t)B_SZ * H_SZ;
    dim3 grid(B_SZ / BLOCK_M, H_SZ / BLOCK_N); // 64 x 16
    lstm_cell_wmma<<<grid, dim3(THREADS), 0, stream>>>(
        xb, hb, wxb, whb, bx, bh, c1, outh, outc);
}
